// SelfAttn_52364241273613
// MI455X (gfx1250) — matrix operand order repaired and checked by reading, not compiled, not run on hardware
//
#include <hip/hip_runtime.h>
#include <hip/hip_bf16.h>

// ---------------------------------------------------------------------------
// Fused self-attention (SAGAN-style) for MI455X / gfx1250, wave32 + WMMA.
//   f = Wf x + bf ; g = Wg x + bg ; h = Wh x + bh        (f16 intermediates)
//   s = f^T g ; attn = softmax_i(s) ; o = h attn ; y = gamma*o + x
// Flash-style streaming over keys (the 4096x4096 attention matrix is never
// materialized; saves ~768MB HBM round trips).  GEMMs use
// v_wmma_f32_16x16x32_f16 (K=32 == C8 so one WMMA = one full S tile).
// The value tile h[256 x 32keys] is staged into double-buffered LDS by the
// Tensor Data Mover (tensor_load_to_lds + s_wait_tensorcnt), shared by all
// 4 waves of the workgroup (4x less VMEM traffic than direct loads).
// ---------------------------------------------------------------------------

#define BATCH 4
#define CCH   256
#define C8CH  32
#define NPIX  4096   // 64*64

typedef __attribute__((ext_vector_type(16))) _Float16     v16h;
typedef __attribute__((ext_vector_type(8)))  _Float16     v8h;
typedef __attribute__((ext_vector_type(8)))  float        v8f;
typedef __attribute__((ext_vector_type(4)))  unsigned int u32x4;
typedef __attribute__((ext_vector_type(4)))  int          i32x4;
typedef __attribute__((ext_vector_type(8)))  int          i32x8;

__device__ __forceinline__ v8f zero8() {
    v8f z;
    #pragma unroll
    for (int i = 0; i < 8; ++i) z[i] = 0.f;
    return z;
}

// A-operand (16x32 f16) loader.  Contiguous K order used here (B uses the same): lane holds row m = lane%16;
// lane-half h0 = lane/16 selects K range [16*h0, 16*h0+16).
// `row` points at this lane's contiguous, K-major row (global or LDS).
__device__ __forceinline__ v16h load_a16(const _Float16* row, int h0) {
    v8h lo = *(const v8h*)(row + 16 * h0);       // K = 16*h0 + 0..7
    v8h hi = *(const v8h*)(row + 16 * h0 + 8);   // K = 16*h0 + 8..15
    v16h a;
    #pragma unroll
    for (int t = 0; t < 8; ++t) { a[t] = lo[t]; a[8 + t] = hi[t]; }
    return a;
}

// ---------------------------------------------------------------------------
// Tensor Data Mover: 2-D tile load  h[0:256, i0:i0+32] (f16) -> LDS.
// D# built per CDNA5 ISA:
//   group0: count=1 | lds_addr | global_addr(57b) | type=2
//   group1: data_size=2B | tensor_dim0=NPIX | tensor_dim1=CCH |
//           tile_dim0=32 | tile_dim1=256 | tensor_dim0_stride=NPIX
// ---------------------------------------------------------------------------
__device__ __forceinline__ void tdm_load_h_tile(unsigned lds_addr,
                                                unsigned long long gaddr) {
    u32x4 g0;
    g0[0] = 1u;                                              // count=1 (valid)
    g0[1] = lds_addr;                                        // LDS byte address
    g0[2] = (unsigned)(gaddr & 0xFFFFFFFFull);               // global_addr[31:0]
    g0[3] = (unsigned)((gaddr >> 32) & 0x01FFFFFFull)        // global_addr[56:32]
          | (2u << 30);                                      // type = 2 ("image")
    i32x8 g1;
    g1[0] = (int)(1u << 16);                                 // data_size=1 -> 2 bytes
    g1[1] = (int)((NPIX & 0xFFFFu) << 16);                   // tensor_dim0 lo16
    g1[2] = (int)(((NPIX >> 16) & 0xFFFFu)                   // tensor_dim0 hi16
          | ((CCH & 0xFFFFu) << 16));                        // tensor_dim1 lo16
    g1[3] = (int)(((CCH >> 16) & 0xFFFFu)                    // tensor_dim1 hi16
          | (32u << 16));                                    // tile_dim0 = 32
    g1[4] = (int)(CCH & 0xFFFFu);                            // tile_dim1=256, tile_dim2=0
    g1[5] = (int)NPIX;                                       // tensor_dim0_stride lo32
    g1[6] = 0;                                               // stride0 hi / stride1 lo
    g1[7] = 0;
    i32x4 z4 = {0, 0, 0, 0};
#if __has_include(<hip/amd_detail/amd_gfx1250_TDM.h>)
    i32x8 z8 = {0, 0, 0, 0, 0, 0, 0, 0};                     // clang-23 6-arg form
    __builtin_amdgcn_tensor_load_to_lds(g0, g1, z4, z4, z8, 0);
#else
    __builtin_amdgcn_tensor_load_to_lds(g0, g1, z4, z4, 0);  // ROCm 7.2 5-arg form
#endif
}

// ---------------------------------------------------------------------------
// Kernel 1: 1x1-conv projection GEMM  out[m,n] = sum_c W[m,c]*x[b,c,n] + bias[m]
// grid = (NPIX/64, M/16, BATCH), block = 128 (4 waves, 16 n-columns each).
// TRANSPOSED=1 stores out[n*M + m] (fT/gT), else out[m*NPIX + n] (h).
// ---------------------------------------------------------------------------
template <int TRANSPOSED>
__launch_bounds__(128, 2)
__global__ void proj_wmma_kernel(const float* __restrict__ x,
                                 const float* __restrict__ W,
                                 const float* __restrict__ bias,
                                 _Float16* __restrict__ out, int M) {
    const int b    = blockIdx.z;
    const int lane = threadIdx.x & 31;
    const int wave = threadIdx.x >> 5;
    const int h0   = lane >> 4;
    const int ln   = lane & 15;
    const int n0   = blockIdx.x * 64 + wave * 16;
    const int m0   = blockIdx.y * 16;

    const float* xb = x + (size_t)b * CCH * NPIX;
    _Float16* outb  = out + (size_t)b * M * NPIX;

    v8f acc = zero8();
    for (int k0 = 0; k0 < CCH; k0 += 32) {
        // A tile: W rows (row-major [M,C] f32), convert to f16 on load.
        const float* wr = W + (size_t)(m0 + ln) * CCH + k0;
        v16h a;
        #pragma unroll
        for (int t = 0; t < 8; ++t) {
            a[t]     = (_Float16)wr[16 * h0 + t];
            a[8 + t] = (_Float16)wr[16 * h0 + 8 + t];
        }
        // B tile: x column gather.  b[idx] = elem(k = 16*h0+idx, n = ln).
        // For fixed k, lanes 0..15 read a contiguous 64B burst.
        v16h bb;
        #pragma unroll
        for (int t = 0; t < 16; ++t)
            bb[t] = (_Float16)xb[(size_t)(k0 + 16 * h0 + t) * NPIX + (n0 + ln)];
        acc = __builtin_amdgcn_wmma_f32_16x16x32_f16(false, a, false, bb,
                                                     (short)0, acc, false, false);
    }
    // C/D layout: VGPR r -> m = m0 + r + 8*h0, n = n0 + ln.
    #pragma unroll
    for (int r = 0; r < 8; ++r) {
        int m = m0 + r + 8 * h0;
        int n = n0 + ln;
        float v = acc[r] + bias[m];
        if (TRANSPOSED) outb[(size_t)n * M + m]    = (_Float16)v;
        else            outb[(size_t)m * NPIX + n] = (_Float16)v;
    }
}

// ---------------------------------------------------------------------------
// Kernel 2: fused flash attention + residual.
// grid = (NPIX/64, BATCH), block = 128.  Each wave owns 16 query columns j and
// the full C=256 value dim (16 accumulator tiles = 128 VGPRs of f32 output).
// Softmax stats for column j live (replicated) in lane pair (j%16, j%16+16).
// Value tiles stream through LDS via TDM, double buffered.
// ---------------------------------------------------------------------------
__launch_bounds__(128, 1)
__global__ void attn_flash_kernel(const _Float16* __restrict__ fT,   // [B][N][32]
                                  const _Float16* __restrict__ gT,   // [B][N][32]
                                  const _Float16* __restrict__ hF,   // [B][C][N]
                                  const float* __restrict__ x,
                                  const float* __restrict__ gamma,
                                  float* __restrict__ y) {
    __shared__ _Float16 hbuf[2][CCH * 32];   // 2 x 16KB double buffer

    const int b    = blockIdx.y;
    const int lane = threadIdx.x & 31;
    const int wave = threadIdx.x >> 5;
    const int h0   = lane >> 4;
    const int ln   = lane & 15;
    const int j    = blockIdx.x * 64 + wave * 16 + ln;   // this lane's query col

    const _Float16* fTb = fT + (size_t)b * NPIX * C8CH;
    const _Float16* gTb = gT + (size_t)b * NPIX * C8CH;
    const _Float16* hb  = hF + (size_t)b * CCH * NPIX;

    // LDS byte offsets of the two h buffers (ptrtoint of LDS addr = offset).
    const unsigned ldsb0 = (unsigned)(size_t)(&hbuf[0][0]);
    const unsigned ldsb1 = (unsigned)(size_t)(&hbuf[1][0]);
    const bool issuer = (wave == 0);

    // B operand for S: g columns of this wave's queries, K = C8 = 32.
    // B layout: lane = n, b[idx] = elem(k = 16*h0 + idx, n) -> contiguous 32B.
    v16h bg = *(const v16h*)(gTb + (size_t)j * C8CH + 16 * h0);

    v8f acc[16];
    #pragma unroll
    for (int t = 0; t < 16; ++t) acc[t] = zero8();

    float m_run = -1e30f;
    float l_run = 0.f;

    // prologue: TDM fetch of key block 0's value tile
    if (issuer) tdm_load_h_tile(ldsb0, (unsigned long long)(size_t)hb);

    for (int blk = 0; blk < NPIX / 32; ++blk) {
        const int i0 = blk * 32;
        if (issuer) __builtin_amdgcn_s_wait_tensorcnt(0);  // tile `blk` landed
        __syncthreads();                                   // visible to all waves
        if (issuer && blk + 1 < NPIX / 32)                 // overlap next DMA
            tdm_load_h_tile((blk & 1) ? ldsb0 : ldsb1,
                            (unsigned long long)(size_t)(hb + i0 + 32));
        const _Float16* hblk = &hbuf[blk & 1][0];

        if (i0 + 32 < NPIX)   // prefetch next fT rows (global_prefetch_b8)
            __builtin_prefetch(fTb + (size_t)(i0 + 32 + ln) * C8CH, 0, 3);

        // ---- S tiles: S[i,j] = sum_c fT[i][c] * g[c][j] (one WMMA each) ----
        v16h a0 = load_a16(fTb + (size_t)(i0 + ln) * C8CH, h0);
        v16h a1 = load_a16(fTb + (size_t)(i0 + 16 + ln) * C8CH, h0);
        v8f s0 = __builtin_amdgcn_wmma_f32_16x16x32_f16(false, a0, false, bg,
                                                        (short)0, zero8(), false, false);
        v8f s1 = __builtin_amdgcn_wmma_f32_16x16x32_f16(false, a1, false, bg,
                                                        (short)0, zero8(), false, false);

        // ---- online softmax over the key axis for this column ----
        float bm = -1e30f;
        #pragma unroll
        for (int r = 0; r < 8; ++r) bm = fmaxf(bm, fmaxf(s0[r], s1[r]));
        bm = fmaxf(bm, __shfl_xor(bm, 16, 32));        // combine lane pair
        float m_new = fmaxf(m_run, bm);
        float alpha = __expf(m_run - m_new);

        float e0[8], e1[8], ssum = 0.f;
        #pragma unroll
        for (int r = 0; r < 8; ++r) {
            e0[r] = __expf(s0[r] - m_new);
            e1[r] = __expf(s1[r] - m_new);
            ssum += e0[r] + e1[r];
        }
        l_run = l_run * alpha + ssum + __shfl_xor(ssum, 16, 32);
        m_run = m_new;

        #pragma unroll
        for (int t = 0; t < 16; ++t) {
            #pragma unroll
            for (int r = 0; r < 8; ++r) acc[t][r] *= alpha;
        }

        // ---- repack P (C/D layout) into B layout for the o-GEMM ----
        // target: b[idx] = P[key = i0 + 16*h0 + idx, col = ln]; own VGPRs hold
        // rows r+8*h0 of each tile -> exchange the other 8 rows via xor-16.
        v16h pv;
        #pragma unroll
        for (int r = 0; r < 8; ++r) {
            float send = h0 ? e0[r] : e1[r];
            float recv = __shfl_xor(send, 16, 32);
            float blo  = h0 ? recv : e0[r];   // rows 16*h0 + 0..7
            float bhi  = h0 ? e1[r] : recv;   // rows 16*h0 + 8..15
            pv[r]     = (_Float16)blo;
            pv[8 + r] = (_Float16)bhi;
        }

        // ---- o[c,j] += h[c, i0..i0+31] * P  (16 c-tiles from LDS, K=32) ----
        #pragma unroll
        for (int ct = 0; ct < 16; ++ct) {
            v16h ah = load_a16(hblk + (ct * 16 + ln) * 32, h0);
            acc[ct] = __builtin_amdgcn_wmma_f32_16x16x32_f16(false, ah, false, pv,
                                                             (short)0, acc[ct],
                                                             false, false);
        }
        __syncthreads();   // all waves done reading buf[blk&1] before reuse
    }

    // ---- epilogue: y = gamma * (o / l) + x ----
    const float inv_l = 1.f / l_run;
    const float gm = gamma[0];
    const float* xb = x + (size_t)b * CCH * NPIX;
    float* yb = y + (size_t)b * CCH * NPIX;
    #pragma unroll
    for (int ct = 0; ct < 16; ++ct) {
        #pragma unroll
        for (int r = 0; r < 8; ++r) {
            int c = ct * 16 + r + 8 * h0;
            size_t idx = (size_t)c * NPIX + j;
            yb[idx] = gm * (acc[ct][r] * inv_l) + xb[idx];
        }
    }
}

// ---------------------------------------------------------------------------
extern "C" void kernel_launch(void* const* d_in, const int* in_sizes, int n_in,
                              void* d_out, int out_size, void* d_ws, size_t ws_size,
                              hipStream_t stream) {
    const float* x     = (const float*)d_in[0];
    const float* Wf    = (const float*)d_in[1];
    const float* bf    = (const float*)d_in[2];
    const float* Wg    = (const float*)d_in[3];
    const float* bg    = (const float*)d_in[4];
    const float* Wh    = (const float*)d_in[5];
    const float* bh    = (const float*)d_in[6];
    const float* gamma = (const float*)d_in[7];
    float* y = (float*)d_out;

    // workspace: fT | gT | h   (all f16, 64B-aligned regions)
    _Float16* fT = (_Float16*)d_ws;                       // B*N*32  (1 MB)
    _Float16* gT = fT + (size_t)BATCH * NPIX * C8CH;      // B*N*32  (1 MB)
    _Float16* hF = gT + (size_t)BATCH * NPIX * C8CH;      // B*C*N   (8 MB)

    dim3 blk(128);
    // projections
    proj_wmma_kernel<1><<<dim3(NPIX / 64, C8CH / 16, BATCH), blk, 0, stream>>>(
        x, Wf, bf, fT, C8CH);
    proj_wmma_kernel<1><<<dim3(NPIX / 64, C8CH / 16, BATCH), blk, 0, stream>>>(
        x, Wg, bg, gT, C8CH);
    proj_wmma_kernel<0><<<dim3(NPIX / 64, CCH / 16, BATCH), blk, 0, stream>>>(
        x, Wh, bh, hF, CCH);
    // fused attention + residual
    attn_flash_kernel<<<dim3(NPIX / 64, BATCH), blk, 0, stream>>>(
        fT, gT, hF, x, gamma, y);
}
